// DynamicGNN_Embedding_GlobalFeatures_13262859010605
// MI455X (gfx1250) — compile-verified, matrix operand order checked by reading, not run on hardware
//
#include <hip/hip_runtime.h>
#include <hip/hip_bf16.h>

typedef __attribute__((ext_vector_type(16))) _Float16 v16h;
typedef __attribute__((ext_vector_type(8)))  _Float16 v8h;
typedef __attribute__((ext_vector_type(4)))  _Float16 v4h;
typedef __attribute__((ext_vector_type(8)))  float    v8f;

#define N_NODES 30000
#define N_EDGES 480000
#define F_IN    17
#define EMB     16
#define D_IN    32
#define H_DIM   256
#define N_GR    64
#define GF_DIM  32
#define P_DIM   128
#define KC_DIM  288   // H + GF

// ---------------------------------------------------------------- utilities

__global__ void zero_f32_kernel(float* p, int n) {
    for (int i = blockIdx.x * blockDim.x + threadIdx.x; i < n; i += gridDim.x * blockDim.x)
        p[i] = 0.0f;
}

// Pack W[K,Nw] (f32, row-major) into WMMA B-fragment order (f16):
// Bp[ ((kt*NT + nt)*32 + lane)*16 + i ] = W[k, n]  with the 16x16x32 f16 striping
__global__ void pack_b_kernel(const float* __restrict__ W, _Float16* __restrict__ Bp,
                              int K, int Nw) {
    int total = K * Nw;
    int NT = Nw >> 4;
    for (int idx = blockIdx.x * blockDim.x + threadIdx.x; idx < total;
         idx += gridDim.x * blockDim.x) {
        int i    = idx & 15;
        int lane = (idx >> 4) & 31;
        int tile = idx >> 9;              // kt*NT + nt
        int nt = tile % NT, kt = tile / NT;
        int k = kt * 32 + (i & 7) + ((lane >> 4) << 3) + ((i >> 3) << 4);
        int n = nt * 16 + (lane & 15);
        Bp[idx] = (_Float16)W[k * Nw + n];
    }
}

// h0[n, 0:16] = x[n, 1:17];  h0[n, 16:32] = emb[int(x[n,0])]
__global__ void build_h0_kernel(const float* __restrict__ x, const float* __restrict__ emb,
                                float* __restrict__ h0) {
    int total = N_NODES * D_IN;
    for (int i = blockIdx.x * blockDim.x + threadIdx.x; i < total; i += gridDim.x * blockDim.x) {
        int n = i >> 5, c = i & 31;
        float v;
        if (c < 16) {
            v = x[n * F_IN + 1 + c];
        } else {
            int ty = (int)x[n * F_IN];
            v = emb[ty * EMB + (c - 16)];
        }
        h0[i] = v;
    }
}

// ---------------------------------------------------------------- degree / norm

__global__ void deg_init_kernel(float* deg) {
    for (int i = blockIdx.x * blockDim.x + threadIdx.x; i < N_NODES; i += gridDim.x * blockDim.x)
        deg[i] = 1.0f;  // self loop
}

__global__ void deg_acc_kernel(const int* __restrict__ dst, float* deg) {
    for (int e = blockIdx.x * blockDim.x + threadIdx.x; e < N_EDGES; e += gridDim.x * blockDim.x)
        atomicAdd(&deg[dst[e]], 1.0f);
}

__global__ void deg_to_dinv_kernel(float* deg) {
    for (int i = blockIdx.x * blockDim.x + threadIdx.x; i < N_NODES; i += gridDim.x * blockDim.x)
        deg[i] = rsqrtf(deg[i]);   // deg >= 1 always (self loop)
}

__global__ void norm_kernel(const int* __restrict__ src, const int* __restrict__ dst,
                            const float* __restrict__ dinv, float* __restrict__ norm) {
    int total = N_EDGES + N_NODES;
    for (int e = blockIdx.x * blockDim.x + threadIdx.x; e < total; e += gridDim.x * blockDim.x) {
        float v;
        if (e < N_EDGES) v = dinv[src[e]] * dinv[dst[e]];
        else { float d = dinv[e - N_EDGES]; v = d * d; }
        norm[e] = v;
    }
}

// ---------------------------------------------------------------- WMMA GEMM
// C[M,NW] = A[M,K](f32) * Bp(pre-swizzled f16 fragments), f32 accumulate.
// M must be a multiple of 16 (true for all call sites). One 16-row M-tile per
// block, 4 waves sweep N-tiles (fully unrolled so A-fragments stay in VGPRs
// across all N-tiles). A tile staged into LDS as f16; each A fragment is two
// contiguous ds_load_b128 + shuffle; B fragments are contiguous 32B loads.
template <int K, int NW, bool RELU, bool HAS_BIAS>
__global__ __launch_bounds__(128)
void wmma_gemm_kernel(const float* __restrict__ A, const _Float16* __restrict__ Bp,
                      const float* __restrict__ bias, float* __restrict__ C) {
    __shared__ _Float16 sA[16 * K];

    const int row0 = blockIdx.x * 16;
    const int tid  = threadIdx.x;

    // cooperative f32->f16 stage of the 16 x K A-tile (float4 loads, b64 LDS stores)
    for (int i = tid * 4; i < 16 * K; i += 128 * 4) {
        int r = i / K, k = i - r * K;
        const float* ap = A + (row0 + r) * K + k;
        float4 f = *(const float4*)ap;
        v4h hv;
        hv[0] = (_Float16)f.x; hv[1] = (_Float16)f.y;
        hv[2] = (_Float16)f.z; hv[3] = (_Float16)f.w;
        *(v4h*)&sA[r * K + k] = hv;
    }
    __syncthreads();

    const int wave = tid >> 5;
    const int lane = tid & 31;
    const int lr   = lane & 15;      // row (A) / col (B/C) within tile
    const int lhi  = lane >> 4;      // half-K selector per ISA striping
    constexpr int NT = NW / 16;
    constexpr int KT = K / 32;

    // A fragments: loop-invariant across N-tiles -> load once into VGPRs
    v16h afr[KT];
#pragma unroll
    for (int kt = 0; kt < KT; ++kt) {
        // lane holds K {base..base+7, base+16..base+23}, base = kt*32 + 8*lhi
        const _Float16* arow = &sA[lr * K + kt * 32 + (lhi << 3)];
        v8h alo = *(const v8h*)(arow);
        v8h ahi = *(const v8h*)(arow + 16);
        afr[kt] = __builtin_shufflevector(alo, ahi,
                    0, 1, 2, 3, 4, 5, 6, 7, 8, 9, 10, 11, 12, 13, 14, 15);
    }

#pragma unroll
    for (int ni = 0; ni < NT / 4; ++ni) {
        const int nt   = ni * 4 + wave;
        const int ncol = nt * 16 + lr;
        v8f acc = {};
#pragma unroll
        for (int kt = 0; kt < KT; ++kt) {
            // B fragment: pre-swizzled, fully contiguous per lane
            v16h bf = *(const v16h*)(Bp + ((size_t)(kt * NT + nt) * 32 + lane) * 16);
            acc = __builtin_amdgcn_wmma_f32_16x16x32_f16(
                false, afr[kt], false, bf, (short)0, acc, false, false);
        }
        float bv = 0.0f;
        if (HAS_BIAS) bv = bias[ncol];
#pragma unroll
        for (int r = 0; r < 8; ++r) {
            float v = acc[r];
            if (HAS_BIAS) v += bv;
            if (RELU) v = fmaxf(v, 0.0f);
            C[(size_t)(row0 + r + (lhi << 3)) * NW + ncol] = v;
        }
    }
}

// ---------------------------------------------------------------- scatter (GCN aggregate)
// one wave per edge (incl. self loops at e >= E); 8 features per lane
__global__ __launch_bounds__(256)
void scatter_kernel(const float* __restrict__ hw, const int* __restrict__ src,
                    const int* __restrict__ dst, const float* __restrict__ norm,
                    float* __restrict__ agg) {
    const int nwaves = gridDim.x * (blockDim.x >> 5);
    const int gw     = blockIdx.x * (blockDim.x >> 5) + (threadIdx.x >> 5);
    const int lane   = threadIdx.x & 31;
    for (int e = gw; e < N_EDGES + N_NODES; e += nwaves) {
        int s, d;
        if (e < N_EDGES) { s = src[e]; d = dst[e]; }
        else             { s = d = e - N_EDGES; }
        float w = norm[e];
        const float* hs = hw  + (size_t)s * H_DIM;
        float*       ad = agg + (size_t)d * H_DIM;
#pragma unroll
        for (int j = 0; j < H_DIM / 32; ++j)
            atomicAdd(&ad[lane + j * 32], w * hs[lane + j * 32]);
    }
}

// ---------------------------------------------------------------- bias + LayerNorm + ReLU
__global__ __launch_bounds__(256)
void ln_relu_kernel(const float* __restrict__ agg, const float* __restrict__ bias,
                    const float* __restrict__ g, const float* __restrict__ be,
                    float* __restrict__ out) {
    __shared__ float s1[256], s2[256];
    int n = blockIdx.x, t = threadIdx.x;
    float v = agg[(size_t)n * H_DIM + t] + bias[t];
    s1[t] = v; s2[t] = v * v;
    __syncthreads();
    for (int off = 128; off > 0; off >>= 1) {
        if (t < off) { s1[t] += s1[t + off]; s2[t] += s2[t + off]; }
        __syncthreads();
    }
    float mu  = s1[0] * (1.0f / H_DIM);
    float var = s2[0] * (1.0f / H_DIM) - mu * mu;
    float r   = rsqrtf(var + 1e-5f);
    float o   = (v - mu) * r * g[t] + be[t];
    out[(size_t)n * H_DIM + t] = fmaxf(o, 0.0f);
}

// ---------------------------------------------------------------- attention gate
__device__ __forceinline__ unsigned enc_f(float f) {
    unsigned u = __float_as_uint(f);
    return (u >> 31) ? ~u : (u | 0x80000000u);
}
__device__ __forceinline__ float dec_f(unsigned e) {
    unsigned u = (e >> 31) ? (e & 0x7FFFFFFFu) : ~e;
    return __uint_as_float(u);
}

__global__ void gmax_init_kernel(unsigned* gmaxU) {
    int i = threadIdx.x;
    if (i < N_GR) gmaxU[i] = enc_f(-__builtin_huge_valf());
}

__global__ __launch_bounds__(256)
void gate_kernel(const float* __restrict__ tmat, const float* __restrict__ gW2,
                 const float* __restrict__ gb2, const int* __restrict__ batch,
                 float* __restrict__ gate, unsigned* __restrict__ gmaxU) {
    int wid  = blockIdx.x * (blockDim.x >> 5) + (threadIdx.x >> 5);
    int lane = threadIdx.x & 31;
    if (wid >= N_NODES) return;
    float s = 0.0f;
#pragma unroll
    for (int j = 0; j < P_DIM / 32; ++j)
        s += tmat[(size_t)wid * P_DIM + lane + j * 32] * gW2[lane + j * 32];
    for (int off = 16; off > 0; off >>= 1) s += __shfl_down(s, off, 32);
    if (lane == 0) {
        float gv = s + gb2[0];
        gate[wid] = gv;
        atomicMax(&gmaxU[batch[wid]], enc_f(gv));
    }
}

__global__ void softmax_e_kernel(const float* __restrict__ gate, const unsigned* __restrict__ gmaxU,
                                 const int* __restrict__ batch, float* __restrict__ ebuf,
                                 float* __restrict__ denom) {
    for (int n = blockIdx.x * blockDim.x + threadIdx.x; n < N_NODES; n += gridDim.x * blockDim.x) {
        int b = batch[n];
        float e = __expf(gate[n] - dec_f(gmaxU[b]));
        ebuf[n] = e;
        atomicAdd(&denom[b], e);
    }
}

__global__ __launch_bounds__(256)
void pooled_kernel(const float* __restrict__ h2, const float* __restrict__ ebuf,
                   const float* __restrict__ denom, const int* __restrict__ batch,
                   float* __restrict__ pooled) {
    int n = blockIdx.x, t = threadIdx.x;
    int b = batch[n];
    float a = ebuf[n] / denom[b];
    atomicAdd(&pooled[b * H_DIM + t], a * h2[(size_t)n * H_DIM + t]);
}

__global__ void build_comb_kernel(const float* __restrict__ pooled,
                                  const float* __restrict__ gfeat, float* __restrict__ comb) {
    int total = N_GR * KC_DIM;
    for (int i = blockIdx.x * blockDim.x + threadIdx.x; i < total; i += gridDim.x * blockDim.x) {
        int g = i / KC_DIM, j = i - g * KC_DIM;
        comb[i] = (j < H_DIM) ? pooled[g * H_DIM + j] : gfeat[g * GF_DIM + (j - H_DIM)];
    }
}

__global__ void final_kernel(const float* __restrict__ t2, const float* __restrict__ cW2,
                             const float* __restrict__ cb2, float* __restrict__ out) {
    int g = threadIdx.x;
    if (g >= N_GR) return;
    float s0 = cb2[0], s1 = cb2[1];
    for (int j = 0; j < P_DIM; ++j) {
        float v = t2[g * P_DIM + j];
        s0 += v * cW2[j * 2 + 0];
        s1 += v * cW2[j * 2 + 1];
    }
    out[g * 2 + 0] = s0;
    out[g * 2 + 1] = s1;
}

// ---------------------------------------------------------------- launch

static inline size_t align256(size_t x) { return (x + 255) & ~(size_t)255; }

extern "C" void kernel_launch(void* const* d_in, const int* in_sizes, int n_in,
                              void* d_out, int out_size, void* d_ws, size_t ws_size,
                              hipStream_t stream) {
    const float* x      = (const float*)d_in[0];
    const int*   ei     = (const int*)  d_in[1];
    const int*   batch  = (const int*)  d_in[2];
    const float* gfeat  = (const float*)d_in[3];
    const float* emb    = (const float*)d_in[4];
    const float* W1     = (const float*)d_in[5];
    const float* b1     = (const float*)d_in[6];
    const float* g1     = (const float*)d_in[7];
    const float* be1    = (const float*)d_in[8];
    const float* W2     = (const float*)d_in[9];
    const float* b2     = (const float*)d_in[10];
    const float* g2     = (const float*)d_in[11];
    const float* be2    = (const float*)d_in[12];
    const float* gW1    = (const float*)d_in[13];
    const float* gb1    = (const float*)d_in[14];
    const float* gW2    = (const float*)d_in[15];
    const float* gb2    = (const float*)d_in[16];
    const float* cW1    = (const float*)d_in[17];
    const float* cb1    = (const float*)d_in[18];
    const float* cW2    = (const float*)d_in[19];
    const float* cb2    = (const float*)d_in[20];
    float* out = (float*)d_out;

    const int* src = ei;            // edge_index[0]
    const int* dst = ei + N_EDGES;  // edge_index[1]

    // ---- workspace carve-up
    char* w = (char*)d_ws;
    size_t off = 0;
    auto take = [&](size_t bytes) { char* p = w + off; off += align256(bytes); return p; };

    float*    h0    = (float*)   take(sizeof(float) * N_NODES * D_IN);
    float*    X     = (float*)   take(sizeof(float) * N_NODES * H_DIM);
    float*    Y     = (float*)   take(sizeof(float) * N_NODES * H_DIM);
    float*    Z     = (float*)   take(sizeof(float) * N_NODES * H_DIM);
    float*    deg   = (float*)   take(sizeof(float) * N_NODES);             // becomes dinv
    float*    norm  = (float*)   take(sizeof(float) * (N_EDGES + N_NODES));
    float*    gate  = (float*)   take(sizeof(float) * N_NODES);
    float*    ebuf  = (float*)   take(sizeof(float) * N_NODES);
    _Float16* W1p   = (_Float16*)take(sizeof(_Float16) * D_IN * H_DIM);
    _Float16* W2p   = (_Float16*)take(sizeof(_Float16) * H_DIM * H_DIM);
    _Float16* gW1p  = (_Float16*)take(sizeof(_Float16) * H_DIM * P_DIM);
    _Float16* cW1p  = (_Float16*)take(sizeof(_Float16) * KC_DIM * P_DIM);
    unsigned* gmaxU = (unsigned*)take(sizeof(unsigned) * N_GR);
    float*    denom = (float*)   take(sizeof(float) * N_GR);
    float*    pooled= (float*)   take(sizeof(float) * N_GR * H_DIM);
    float*    comb  = (float*)   take(sizeof(float) * N_GR * KC_DIM);
    float*    t2    = (float*)   take(sizeof(float) * N_GR * P_DIM);
    (void)ws_size;

    const int TPB = 256;
    const int MT  = N_NODES / 16;           // 1875, exact
    const int SC_BLK = (N_EDGES + N_NODES + 7) / 8;  // 1 wave per edge, 8 waves/block

    // weights -> pre-swizzled f16 B fragments
    pack_b_kernel<<<(D_IN*H_DIM + TPB-1)/TPB,  TPB, 0, stream>>>(W1,  W1p,  D_IN,  H_DIM);
    pack_b_kernel<<<(H_DIM*H_DIM + TPB-1)/TPB, TPB, 0, stream>>>(W2,  W2p,  H_DIM, H_DIM);
    pack_b_kernel<<<(H_DIM*P_DIM + TPB-1)/TPB, TPB, 0, stream>>>(gW1, gW1p, H_DIM, P_DIM);
    pack_b_kernel<<<(KC_DIM*P_DIM + TPB-1)/TPB,TPB, 0, stream>>>(cW1, cW1p, KC_DIM,P_DIM);

    // node features + embedding concat
    build_h0_kernel<<<(N_NODES*D_IN + TPB-1)/TPB, TPB, 0, stream>>>(x, emb, h0);

    // GCN normalization
    deg_init_kernel<<<(N_NODES + TPB-1)/TPB, TPB, 0, stream>>>(deg);
    deg_acc_kernel<<<(N_EDGES + TPB-1)/TPB, TPB, 0, stream>>>(dst, deg);
    deg_to_dinv_kernel<<<(N_NODES + TPB-1)/TPB, TPB, 0, stream>>>(deg);
    norm_kernel<<<(N_EDGES + N_NODES + TPB-1)/TPB, TPB, 0, stream>>>(src, dst, deg, norm);

    // ---- layer 1: hw1 = h0 @ W1  (WMMA, K=32)
    wmma_gemm_kernel<D_IN, H_DIM, false, false><<<MT, 128, 0, stream>>>(h0, W1p, nullptr, X);
    zero_f32_kernel<<<(N_NODES*H_DIM + TPB-1)/TPB, TPB, 0, stream>>>(Y, N_NODES*H_DIM);
    scatter_kernel<<<SC_BLK, 256, 0, stream>>>(X, src, dst, norm, Y);
    ln_relu_kernel<<<N_NODES, 256, 0, stream>>>(Y, b1, g1, be1, X);     // h1 -> X

    // ---- layer 2: hw2 = h1 @ W2  (WMMA, K=256)
    wmma_gemm_kernel<H_DIM, H_DIM, false, false><<<MT, 128, 0, stream>>>(X, W2p, nullptr, Y);
    zero_f32_kernel<<<(N_NODES*H_DIM + TPB-1)/TPB, TPB, 0, stream>>>(Z, N_NODES*H_DIM);
    scatter_kernel<<<SC_BLK, 256, 0, stream>>>(Y, src, dst, norm, Z);
    ln_relu_kernel<<<N_NODES, 256, 0, stream>>>(Z, b2, g2, be2, Y);     // h2 -> Y

    // ---- attention gate: t = relu(h2 @ gW1 + gb1)  (WMMA, K=256, NW=128)
    wmma_gemm_kernel<H_DIM, P_DIM, true, true><<<MT, 128, 0, stream>>>(Y, gW1p, gb1, Z);
    gmax_init_kernel<<<1, 64, 0, stream>>>(gmaxU);
    gate_kernel<<<(N_NODES*32 + 255)/256, 256, 0, stream>>>(Z, gW2, gb2, batch, gate, gmaxU);
    zero_f32_kernel<<<1, TPB, 0, stream>>>(denom, N_GR);
    softmax_e_kernel<<<(N_NODES + TPB-1)/TPB, TPB, 0, stream>>>(gate, gmaxU, batch, ebuf, denom);
    zero_f32_kernel<<<(N_GR*H_DIM + TPB-1)/TPB, TPB, 0, stream>>>(pooled, N_GR*H_DIM);
    pooled_kernel<<<N_NODES, 256, 0, stream>>>(Y, ebuf, denom, batch, pooled);

    // ---- classifier: relu(comb @ cW1 + cb1) @ cW2 + cb2   (WMMA, M=64, K=288)
    build_comb_kernel<<<(N_GR*KC_DIM + TPB-1)/TPB, TPB, 0, stream>>>(pooled, gfeat, comb);
    wmma_gemm_kernel<KC_DIM, P_DIM, true, true><<<N_GR / 16, 128, 0, stream>>>(comb, cW1p, cb1, t2);
    final_kernel<<<1, 64, 0, stream>>>(t2, cW2, cb2, out);
    (void)in_sizes; (void)n_in; (void)out_size;
}
